// KalmanGNN_32298154066140
// MI455X (gfx1250) — compile-verified, matrix operand order checked by reading, not run on hardware
//
#include <hip/hip_runtime.h>

typedef __attribute__((ext_vector_type(16))) _Float16 v16h;
typedef __attribute__((ext_vector_type(8)))  _Float16 v8h;
typedef __attribute__((ext_vector_type(4)))  _Float16 v4h;
typedef __attribute__((ext_vector_type(8)))  float    v8f;
typedef __attribute__((ext_vector_type(4)))  float    v4f;

#define NTOK 4096
#define HD   64
#define BB   64
#define XD   4

struct KParams {
  const float* hx; const float* hy;
  const float* msg[3];                                  // pc, fc, yc messages
  const float* W1[3]; const float* b1v[3];
  const float* W2[3]; const float* b2v[3];
  const float* ndW1; const float* ndb1; const float* ndW2; const float* ndb2;
  const float* Wihp; const float* bihp; const float* Whhp; const float* bhhp;
  const float* dW1; const float* db1p; const float* dW2; const float* db2p;
  float* eps; float* hnew;
};

// ---------------- LDS layout (bytes) ----------------
// All weights stored PACKED per 32x16 WMMA tile: tile*512 halves,
// within tile: chunk c (0/1)*256 + lane*8 + h  (lane = hi*16 + n).
// Encoder W1 uses permuted K' = [side 0..63 | msg 64..67, zeros..95 | hx 96..159]
// => 5 ktiles x 4 ntiles = 20 tiles per encoder.
#define OFF_WENC1 0                                   // 3 * 20 tiles
#define OFF_WENC2 (OFF_WENC1 + 3*20*512*2)            // 3 * 8 tiles
#define OFF_ND1   (OFF_WENC2 + 3*8*512*2)
#define OFF_ND2   (OFF_ND1 + 8*512*2)
#define OFF_WIH   (OFF_ND2 + 8*512*2)                 // 24 tiles
#define OFF_WHH   (OFF_WIH + 24*512*2)
#define OFF_DEC1  (OFF_WHH + 24*512*2)                // 8 tiles
#define OFF_DEC2  (OFF_DEC1 + 8*512*2)                // 2 tiles (N padded 4->16)
#define OFF_BIAS  (OFF_DEC2 + 2*512*2)                // f32 biases
#define BO_ENC1 0
#define BO_ENC2 192
#define BO_ND1  384
#define BO_ND2  448
#define BO_IH   512
#define BO_HH   704
#define BO_D1   896
#define BO_D2   960
#define NBIAS   976
#define OFF_WAVE (OFF_BIAS + NBIAS*4)
// per-wave: regA 4352B (bufA f16[16][104] / reused f32 hnT[16][68]),
//           hxT f32[16][68] 4352B, actb f16[16][72] 2304B
#define WAVE_BYTES (4352 + 4352 + 2304)
#define LDS_TOTAL  (OFF_WAVE + 8*WAVE_BYTES)          // = 253760 B < 320 KB

// ---------------- fragment helpers ----------------
// 16-bit fragment K map: half h (0..15) -> K = (h<8 ? h : h+8) + 8*hi
// => fragment = two contiguous 8-half chunks at K = kbase+8*hi and kbase+16+8*hi.

__device__ __forceinline__ v16h cat8(v8h lo, v8h hp) {
  return __builtin_shufflevector(lo, hp, 0,1,2,3,4,5,6,7,8,9,10,11,12,13,14,15);
}

__device__ __forceinline__ v16h load_a16v(const _Float16* buf, int strideH, int kbase, int m, int hi) {
  const _Float16* p = buf + m*strideH + kbase + hi*8;
  v8h lo = *(const v8h*)p;
  v8h hp = *(const v8h*)(p + 16);
  return cat8(lo, hp);
}

__device__ __forceinline__ v16h load_a32v(const float* buf, int strideF, int kbase, int m, int hi) {
  const float* p = buf + m*strideF + kbase + hi*8;
  v4f f0 = *(const v4f*)p;
  v4f f1 = *(const v4f*)(p + 4);
  v4f f2 = *(const v4f*)(p + 16);
  v4f f3 = *(const v4f*)(p + 20);
  v16h a;
#pragma unroll
  for (int j = 0; j < 4; ++j) {
    a[j]      = (_Float16)f0[j];
    a[4 + j]  = (_Float16)f1[j];
    a[8 + j]  = (_Float16)f2[j];
    a[12 + j] = (_Float16)f3[j];
  }
  return a;
}

__device__ __forceinline__ v16h load_bp(const _Float16* packed, int tile, int lane) {
  const _Float16* p = packed + tile*512 + lane*8;
  v8h lo = *(const v8h*)p;
  v8h hp = *(const v8h*)(p + 256);
  return cat8(lo, hp);
}

__device__ __forceinline__ v8f wmma16(v16h a, v16h b, v8f c) {
  return __builtin_amdgcn_wmma_f32_16x16x32_f16(false, a, false, b, (short)0, c, false, false);
}

__device__ __forceinline__ void bias_relu4(v8f* c, const float* bias, int nl) {
#pragma unroll
  for (int nt = 0; nt < 4; ++nt)
#pragma unroll
    for (int v = 0; v < 8; ++v)
      c[nt][v] = fmaxf(c[nt][v] + bias[nt*16 + nl], 0.f);
}

__device__ __forceinline__ void store_c16(_Float16* act, int stride, const v8f* c, int nl, int hi) {
#pragma unroll
  for (int nt = 0; nt < 4; ++nt)
#pragma unroll
    for (int v = 0; v < 8; ++v)
      act[(v + 8*hi)*stride + nt*16 + nl] = (_Float16)c[nt][v];
}

// stage weight [K][Nout] (f32, row-major) into packed B-fragment tiles (f16)
__device__ __forceinline__ void stage_packedB(_Float16* dst, const float* src,
                                              int K, int Nout, int tid) {
  const int ntiles = Nout >> 4;
  const int total = K * Nout;
  for (int i = tid; i < total; i += 256) {
    int n  = i & 15;
    int nt = (i >> 4) % ntiles;
    int k  = i / Nout;
    int kt = k >> 5, kin = k & 31;
    int c  = kin >> 4, hh = (kin >> 3) & 1, h = kin & 7;
    dst[(kt*ntiles + nt)*512 + c*256 + (hh*16 + n)*8 + h] = (_Float16)src[i];
  }
}

// dec_W2 64x4 -> single padded 16-col tile group
__device__ __forceinline__ void stage_packedB_pad(_Float16* dst, const float* src,
                                                  int K, int srcN, int tid) {
  const int total = K * 16;
  for (int i = tid; i < total; i += 256) {
    int n = i & 15;
    int k = i >> 4;
    int kt = k >> 5, kin = k & 31;
    int c = kin >> 4, hh = (kin >> 3) & 1, h = kin & 7;
    dst[kt*512 + c*256 + (hh*16 + n)*8 + h] =
        (n < srcN) ? (_Float16)src[k*srcN + n] : (_Float16)0.f;
  }
}

__device__ __forceinline__ float sigm(float x) { return 1.f / (1.f + __expf(-x)); }

// ---------------- kernel ----------------
__global__ void __launch_bounds__(256) kgnn_wmma_kernel(KParams P) {
  extern __shared__ char smem[];
  const int tid  = threadIdx.x;
  const int lane = tid & 31;
  const int w    = tid >> 5;
  const int hi   = lane >> 4;
  const int nl   = lane & 15;    // A-row (M) and B/C column (N) for this lane

  _Float16* Wenc1 = (_Float16*)(smem + OFF_WENC1);
  _Float16* Wenc2 = (_Float16*)(smem + OFF_WENC2);
  _Float16* Nd1   = (_Float16*)(smem + OFF_ND1);
  _Float16* Nd2   = (_Float16*)(smem + OFF_ND2);
  _Float16* Wih   = (_Float16*)(smem + OFF_WIH);
  _Float16* Whh   = (_Float16*)(smem + OFF_WHH);
  _Float16* Dc1   = (_Float16*)(smem + OFF_DEC1);
  _Float16* Dc2   = (_Float16*)(smem + OFF_DEC2);
  float*    Bias  = (float*)(smem + OFF_BIAS);

  // ---- stage all weights into LDS as packed f16 fragment tiles ----
  // Encoder W1: permuted K' (side rows k<64 -> k' = k ; hx rows 64..127 -> k' = k+32).
  for (int e = 0; e < 3; ++e) {
    const float* src = P.W1[e];
    _Float16* dst = Wenc1 + e*(20*512);
    for (int i = tid; i < 128*64; i += 256) {
      int n = i & 15;
      int nt = (i >> 4) & 3;
      int k = i >> 6;
      int kp = (k < 64) ? k : k + 32;
      int kt = kp >> 5, kin = kp & 31;
      int c = kin >> 4, hh = (kin >> 3) & 1, h = kin & 7;
      dst[(kt*4 + nt)*512 + c*256 + (hh*16 + n)*8 + h] = (_Float16)src[i];
    }
  }
  // Encoder W1 ktile 2 (K' 64..95): msg rows 128..131 at K'=64..67, zeros elsewhere.
  for (int i = tid; i < 3*4*512; i += 256) {
    int e  = i >> 11;             // / 2048
    int r  = i & 2047;
    int nt = r >> 9;
    int q  = r & 511;
    int c = q >> 8, ln = (q >> 3) & 31, h = q & 7;
    _Float16 val = (_Float16)0.f;
    if (c == 0 && ln < 16 && h < 4)          // K' = 64+h  (hi=0 chunk0)
      val = (_Float16)P.W1[e][(128 + h)*64 + nt*16 + ln];
    Wenc1[e*(20*512) + (8 + nt)*512 + c*256 + ln*8 + h] = val;
  }
  for (int e = 0; e < 3; ++e)
    stage_packedB(Wenc2 + e*(8*512), P.W2[e], 64, 64, tid);
  stage_packedB(Nd1, P.ndW1, 64, 64, tid);
  stage_packedB(Nd2, P.ndW2, 64, 64, tid);
  stage_packedB(Wih, P.Wihp, 64, 192, tid);
  stage_packedB(Whh, P.Whhp, 64, 192, tid);
  stage_packedB(Dc1, P.dW1, 64, 64, tid);
  stage_packedB_pad(Dc2, P.dW2, 64, 4, tid);
  for (int e = 0; e < 3; ++e)
    for (int i = tid; i < 64; i += 256) {
      Bias[BO_ENC1 + e*64 + i] = P.b1v[e][i];
      Bias[BO_ENC2 + e*64 + i] = P.b2v[e][i];
    }
  for (int i = tid; i < 64; i += 256) {
    Bias[BO_ND1 + i] = P.ndb1[i];
    Bias[BO_ND2 + i] = P.ndb2[i];
    Bias[BO_D1  + i] = P.db1p[i];
  }
  for (int i = tid; i < 192; i += 256) {
    Bias[BO_IH + i] = P.bihp[i];
    Bias[BO_HH + i] = P.bhhp[i];
  }
  for (int i = tid; i < 16; i += 256) Bias[BO_D2 + i] = (i < 4) ? P.db2p[i] : 0.f;
  __syncthreads();

  // ---- per-wave tile of 16 tokens ----
  char*     wbase = smem + OFF_WAVE + w*WAVE_BYTES;
  _Float16* bufA  = (_Float16*)wbase;             // [16][104] f16 side input + msg + pad
  float*    hnT   = (float*)wbase;                // [16][68] f32 hnew (reuses bufA)
  float*    hxT   = (float*)(wbase + 4352);       // [16][68] f32 hx tile
  _Float16* actb  = (_Float16*)(wbase + 8704);    // [16][72] f16 activations

  const int tok0   = blockIdx.x*128 + w*16;
  const int bb     = tok0 >> 12;                  // N = 4096
  const int n0     = tok0 & (NTOK - 1);
  const int hxBase = bb*HD*NTOK;

  // stage hx tile [token][feature] f32 (vector global loads, scalar b32 transpose)
  for (int i = lane; i < 256; i += 32) {
    int h = i >> 2, tq = (i & 3) * 4;
    v4f d = *(const v4f*)&P.hx[hxBase + h*NTOK + n0 + tq];
#pragma unroll
    for (int j = 0; j < 4; ++j) hxT[(tq + j)*68 + h] = d[j];
  }
  // zero bufA cols 64..95 (msg cols rewritten per encoder; 68..95 stay zero)
  {
    v8h z8 = {};
    for (int i = lane; i < 64; i += 32) {
      int t = i >> 2, q = (i & 3) * 8;
      *(v8h*)&bufA[t*104 + 64 + q] = z8;
    }
  }

  v8f sum[4] = { {}, {}, {}, {} };

  for (int e = 0; e < 3; ++e) {
    // ---- side input cols 0..63 ----
    if (e < 2) {
      const int sh = (e == 0) ? -1 : 1;
      for (int i = lane; i < 256; i += 32) {
        int t = i >> 4, hq = (i & 15) * 4;
        int ts = t + sh;
        v4f d;
        if (ts >= 0 && ts <= 15) {
          d = *(const v4f*)&hxT[ts*68 + hq];
        } else {
          int nn = n0 + ts;
          nn = nn < 0 ? 0 : (nn > NTOK - 1 ? NTOK - 1 : nn);
#pragma unroll
          for (int j = 0; j < 4; ++j) d[j] = P.hx[hxBase + (hq + j)*NTOK + nn];
        }
        v4h o;
#pragma unroll
        for (int j = 0; j < 4; ++j) o[j] = (_Float16)d[j];
        *(v4h*)&bufA[t*104 + hq] = o;
      }
    } else {
      for (int i = lane; i < 256; i += 32) {
        int h = i >> 2, tq = (i & 3) * 4;
        v4f d = *(const v4f*)&P.hy[hxBase + h*NTOK + n0 + tq];
#pragma unroll
        for (int j = 0; j < 4; ++j) bufA[(tq + j)*104 + h] = (_Float16)d[j];
      }
    }
    // msg features at K' 64..67
    const float* mg = P.msg[e];
    for (int i = lane; i < 64; i += 32) {
      int x = i >> 4, t = i & 15;
      bufA[t*104 + 64 + x] = (_Float16)mg[bb*XD*NTOK + x*NTOK + n0 + t];
    }

    // ---- layer 1: K' = 160 (side | msg+pad | hx), all in WMMA ----
    v8f acc[4] = { {}, {}, {}, {} };
    const _Float16* W1p = Wenc1 + e*(20*512);
    v16h aA[5];
    aA[0] = load_a16v(bufA, 104, 0, nl, hi);
    aA[1] = load_a16v(bufA, 104, 32, nl, hi);
    aA[2] = load_a16v(bufA, 104, 64, nl, hi);
    aA[3] = load_a32v(hxT, 68, 0, nl, hi);
    aA[4] = load_a32v(hxT, 68, 32, nl, hi);
#pragma unroll
    for (int ks = 0; ks < 5; ++ks)
#pragma unroll
      for (int nt = 0; nt < 4; ++nt)
        acc[nt] = wmma16(aA[ks], load_bp(W1p, ks*4 + nt, lane), acc[nt]);
    bias_relu4(acc, Bias + BO_ENC1 + e*64, nl);
    store_c16(actb, 72, acc, nl, hi);

    // ---- layer 2: 16x64 @ 64x64 ----
    v8f acc2[4] = { {}, {}, {}, {} };
    {
      v16h a0 = load_a16v(actb, 72, 0, nl, hi);
      v16h a1 = load_a16v(actb, 72, 32, nl, hi);
      const _Float16* Wp = Wenc2 + e*(8*512);
#pragma unroll
      for (int nt = 0; nt < 4; ++nt) {
        acc2[nt] = wmma16(a0, load_bp(Wp, nt, lane), acc2[nt]);
        acc2[nt] = wmma16(a1, load_bp(Wp, 4 + nt, lane), acc2[nt]);
      }
    }
    bias_relu4(acc2, Bias + BO_ENC2 + e*64, nl);
#pragma unroll
    for (int nt = 0; nt < 4; ++nt) sum[nt] += acc2[nt];
  }

  // ---- node MLP on (pe + fe + ye) ----
  store_c16(actb, 72, sum, nl, hi);
  v8f u1[4] = { {}, {}, {}, {} };
  {
    v16h a0 = load_a16v(actb, 72, 0, nl, hi);
    v16h a1 = load_a16v(actb, 72, 32, nl, hi);
#pragma unroll
    for (int nt = 0; nt < 4; ++nt) {
      u1[nt] = wmma16(a0, load_bp(Nd1, nt, lane), u1[nt]);
      u1[nt] = wmma16(a1, load_bp(Nd1, 4 + nt, lane), u1[nt]);
    }
  }
  bias_relu4(u1, Bias + BO_ND1, nl);
  store_c16(actb, 72, u1, nl, hi);
  v8f u2[4] = { {}, {}, {}, {} };
  {
    v16h a0 = load_a16v(actb, 72, 0, nl, hi);
    v16h a1 = load_a16v(actb, 72, 32, nl, hi);
#pragma unroll
    for (int nt = 0; nt < 4; ++nt) {
      u2[nt] = wmma16(a0, load_bp(Nd2, nt, lane), u2[nt]);
      u2[nt] = wmma16(a1, load_bp(Nd2, 4 + nt, lane), u2[nt]);
    }
  }
  bias_relu4(u2, Bias + BO_ND2, nl);
  store_c16(actb, 72, u2, nl, hi);                // U lives in actb

  // ---- GRU cell: gi = U@W_ih, gh = hprev@W_hh (chunked by 16 cols) ----
  v16h aU0 = load_a16v(actb, 72, 0, nl, hi);
  v16h aU1 = load_a16v(actb, 72, 32, nl, hi);
  v16h aH0 = load_a32v(hxT, 68, 0, nl, hi);
  v16h aH1 = load_a32v(hxT, 68, 32, nl, hi);
#pragma unroll
  for (int f = 0; f < 4; ++f) {
    v8f gr = {}, gz = {}, gg = {}, hr = {}, hz = {}, hg = {};
    gr = wmma16(aU0, load_bp(Wih, f,      lane), gr);
    gr = wmma16(aU1, load_bp(Wih, 12 + f, lane), gr);
    gz = wmma16(aU0, load_bp(Wih, 4 + f,  lane), gz);
    gz = wmma16(aU1, load_bp(Wih, 16 + f, lane), gz);
    gg = wmma16(aU0, load_bp(Wih, 8 + f,  lane), gg);
    gg = wmma16(aU1, load_bp(Wih, 20 + f, lane), gg);
    hr = wmma16(aH0, load_bp(Whh, f,      lane), hr);
    hr = wmma16(aH1, load_bp(Whh, 12 + f, lane), hr);
    hz = wmma16(aH0, load_bp(Whh, 4 + f,  lane), hz);
    hz = wmma16(aH1, load_bp(Whh, 16 + f, lane), hz);
    hg = wmma16(aH0, load_bp(Whh, 8 + f,  lane), hg);
    hg = wmma16(aH1, load_bp(Whh, 20 + f, lane), hg);
    const int col = f*16 + nl;
    const float bir = Bias[BO_IH + col],       bhr = Bias[BO_HH + col];
    const float biz = Bias[BO_IH + 64 + col],  bhz = Bias[BO_HH + 64 + col];
    const float big = Bias[BO_IH + 128 + col], bhg = Bias[BO_HH + 128 + col];
    v4f o0, o1;
#pragma unroll
    for (int v = 0; v < 8; ++v) {
      int t = v + 8*hi;
      float r  = sigm(gr[v] + bir + hr[v] + bhr);
      float z  = sigm(gz[v] + biz + hz[v] + bhz);
      float ag = gg[v] + big + r*(hg[v] + bhg);
      float ng = 2.f*sigm(2.f*ag) - 1.f;            // tanh(ag)
      float hp = hxT[t*68 + col];
      float hn = (1.f - z)*ng + z*hp;
      hnT[t*68 + col] = hn;                         // decoder input
      if (v < 4) o0[v] = hn; else o1[v - 4] = hn;
    }
    // direct vector store of hnew: 8 consecutive tokens at feature `col`
    float* po = P.hnew + hxBase + col*NTOK + n0 + 8*hi;
    *(v4f*)po = o0;
    *(v4f*)(po + 4) = o1;
  }

  // ---- decoder: hnew -> 64 (ReLU) -> 4 ----
  v16h d0 = load_a32v(hnT, 68, 0, nl, hi);
  v16h d1 = load_a32v(hnT, 68, 32, nl, hi);
  v8f da[4] = { {}, {}, {}, {} };
#pragma unroll
  for (int nt = 0; nt < 4; ++nt) {
    da[nt] = wmma16(d0, load_bp(Dc1, nt, lane), da[nt]);
    da[nt] = wmma16(d1, load_bp(Dc1, 4 + nt, lane), da[nt]);
  }
  bias_relu4(da, Bias + BO_D1, nl);
  store_c16(actb, 72, da, nl, hi);
  v8f ec = {};
  {
    v16h a0 = load_a16v(actb, 72, 0, nl, hi);
    v16h a1 = load_a16v(actb, 72, 32, nl, hi);
    ec = wmma16(a0, load_bp(Dc2, 0, lane), ec);
    ec = wmma16(a1, load_bp(Dc2, 1, lane), ec);
  }
  if (nl < 4) {
    float bx = Bias[BO_D2 + nl];
    v4f e0, e1;
#pragma unroll
    for (int j = 0; j < 4; ++j) { e0[j] = ec[j] + bx; e1[j] = ec[4 + j] + bx; }
    float* p = P.eps + bb*XD*NTOK + nl*NTOK + n0 + 8*hi;
    *(v4f*)p = e0;
    *(v4f*)(p + 4) = e1;
  }
}

extern "C" void kernel_launch(void* const* d_in, const int* in_sizes, int n_in,
                              void* d_out, int out_size, void* d_ws, size_t ws_size,
                              hipStream_t stream) {
  (void)in_sizes; (void)n_in; (void)out_size; (void)d_ws; (void)ws_size;
  KParams P;
  P.hx     = (const float*)d_in[0];
  P.msg[0] = (const float*)d_in[1];
  P.msg[1] = (const float*)d_in[2];
  P.msg[2] = (const float*)d_in[3];
  P.hy     = (const float*)d_in[4];
  for (int e = 0; e < 3; ++e) {
    P.W1[e]  = (const float*)d_in[5 + 4*e];
    P.b1v[e] = (const float*)d_in[6 + 4*e];
    P.W2[e]  = (const float*)d_in[7 + 4*e];
    P.b2v[e] = (const float*)d_in[8 + 4*e];
  }
  P.ndW1 = (const float*)d_in[17]; P.ndb1 = (const float*)d_in[18];
  P.ndW2 = (const float*)d_in[19]; P.ndb2 = (const float*)d_in[20];
  P.Wihp = (const float*)d_in[21]; P.bihp = (const float*)d_in[22];
  P.Whhp = (const float*)d_in[23]; P.bhhp = (const float*)d_in[24];
  P.dW1  = (const float*)d_in[25]; P.db1p = (const float*)d_in[26];
  P.dW2  = (const float*)d_in[27]; P.db2p = (const float*)d_in[28];
  P.eps  = (float*)d_out;
  P.hnew = (float*)d_out + (size_t)BB*XD*NTOK;

  dim3 grid((BB*NTOK) / 128);     // 2048 blocks, 8 waves (128 tokens) each
  kgnn_wmma_kernel<<<grid, 256, LDS_TOTAL, stream>>>(P);
}
